// PrototypeLayer_55980603736178
// MI455X (gfx1250) — matrix operand order repaired and checked by reading, not compiled, not run on hardware
//
#include <hip/hip_runtime.h>
#include <math.h>

typedef __bf16 bf16_t;
typedef __attribute__((ext_vector_type(16))) __bf16 v16bf;
typedef __attribute__((ext_vector_type(8)))  float  v8f;

#define B_   64
#define D_   128
#define N_   196
#define P_   2000
#define NT_  13      // ceil(196/16)
#define PT_  125     // 2000/16
#define EPS_ 1e-7f

// workspace layout: [psq: 2000 f32, padded to 8192 B][hi frags][lo frags]
#define PSQ_BYTES   8192
#define FRAG_ELEMS  ((size_t)PT_ * 4 * 32 * 16)   // 256000 bf16 per plane
#define WS_NEEDED   (PSQ_BYTES + 2 * FRAG_ELEMS * 2)

// ---------------------------------------------------------------------------
// Prepass 1: exact fp32 prototype norms
__global__ __launch_bounds__(256) void proto_psq_kernel(const float* __restrict__ proto,
                                                        float* __restrict__ psq) {
    int p = blockIdx.x * blockDim.x + threadIdx.x;
    if (p >= P_) return;
    const float4* row = (const float4*)(proto + (size_t)p * D_);
    float s = 0.0f;
    #pragma unroll
    for (int q = 0; q < D_ / 4; ++q) {
        float4 f = row[q];
        s += f.x * f.x + f.y * f.y + f.z * f.z + f.w * f.w;
    }
    psq[p] = s;
}

// Prepass 2: prototypes -> bf16 hi/lo split, stored in WMMA B-fragment layout.
// One wave per (p_tile, k_chunk). B k order used here (same as A): col = lane&15, K = e + 16*(lane>>4).
__global__ __launch_bounds__(256) void proto_frag_kernel(const float* __restrict__ proto,
                                                         bf16_t* __restrict__ whi,
                                                         bf16_t* __restrict__ wlo) {
    int lane = threadIdx.x & 31;
    int task = blockIdx.x * 8 + (threadIdx.x >> 5);   // pt*4 + kc
    if (task >= PT_ * 4) return;
    int pt = task >> 2, kc = task & 3;
    int lanehi = lane >> 4, lanelo = lane & 15;

    const float4* src = (const float4*)(proto + (size_t)(pt * 16 + lanelo) * D_ +
                                        32 * kc + 16 * lanehi);
    v16bf hi, lo;
    #pragma unroll
    for (int q = 0; q < 4; ++q) {
        float4 f4 = src[q];
        float fs[4] = {f4.x, f4.y, f4.z, f4.w};
        #pragma unroll
        for (int c = 0; c < 4; ++c) {
            bf16_t h = (bf16_t)fs[c];
            hi[q * 4 + c] = h;
            lo[q * 4 + c] = (bf16_t)(fs[c] - (float)h);
        }
    }
    size_t off = ((size_t)task * 32 + lane) * 16;     // 32B-aligned per lane
    *(v16bf*)(whi + off) = hi;
    *(v16bf*)(wlo + off) = lo;
}

// ---------------------------------------------------------------------------
// Main kernel (fast path): two p-tiles per iteration, two independent
// accumulator chains interleaved to double matrix-pipe ILP.
__global__ __launch_bounds__(256) void proto_sim_fast(const float* __restrict__ z,
                                                      const float* __restrict__ psq_arr,
                                                      const bf16_t* __restrict__ whi,
                                                      const bf16_t* __restrict__ wlo,
                                                      float* __restrict__ out) {
    __shared__ float zt[D_ * 16];
    __shared__ float zsq[16];

    const int t      = threadIdx.x;
    const int lane   = t & 31;
    const int wid    = t >> 5;
    const int lanehi = lane >> 4;
    const int lanelo = lane & 15;

    const int ntile = blockIdx.x;
    const int b     = blockIdx.y;
    const int n0    = ntile * 16;

    // stage z[b, :, n0:n0+16] (pad 0)
    const float* zb = z + (size_t)b * D_ * N_;
    #pragma unroll
    for (int j = 0; j < 8; ++j) {
        int i = t + 256 * j;
        int d = i >> 4, m = i & 15;
        int n = n0 + m;
        zt[i] = (n < N_) ? zb[d * N_ + n] : 0.0f;
    }
    __syncthreads();

    if (t < 16) {
        float s = 0.0f;
        for (int d = 0; d < D_; ++d) { float v = zt[d * 16 + t]; s += v * v; }
        zsq[t] = s;
    }

    // A fragments (hi/lo) in VGPRs; k order used here (same as B): K = e+16*lanehi
    v16bf ahi[4], alo[4];
    #pragma unroll
    for (int kc = 0; kc < 4; ++kc) {
        #pragma unroll
        for (int e = 0; e < 16; ++e) {
            int K  = e + 16 * lanehi;
            float f = zt[(32 * kc + K) * 16 + lanelo];
            bf16_t h = (bf16_t)f;
            ahi[kc][e] = h;
            alo[kc][e] = (bf16_t)(f - (float)h);
        }
    }
    __syncthreads();

    // hoist z norms; +INF masks padded rows out of the min-reduction for free
    float zsr[8];
    #pragma unroll
    for (int r = 0; r < 8; ++r) {
        int nn = n0 + r + 8 * lanehi;
        zsr[r] = (nn < N_) ? zsq[r + 8 * lanehi] : INFINITY;
    }

    // 125 tiles = 63 pairs (last pair clamps -> duplicate work, identical atomics)
    for (int pp = wid; pp < 63; pp += 8) {
        const int pt0 = 2 * pp;
        const int pt1 = (2 * pp + 1 <= PT_ - 1) ? 2 * pp + 1 : 2 * pp;
        const int pA = pt0 * 16, pB = pt1 * 16;

        float psqA = psq_arr[pA + lanelo];
        float psqB = psq_arr[pB + lanelo];

        const bf16_t* hb0 = whi + ((size_t)(pt0 * 4) * 32 + lane) * 16;
        const bf16_t* lb0 = wlo + ((size_t)(pt0 * 4) * 32 + lane) * 16;
        const bf16_t* hb1 = whi + ((size_t)(pt1 * 4) * 32 + lane) * 16;
        const bf16_t* lb1 = wlo + ((size_t)(pt1 * 4) * 32 + lane) * 16;

        v8f acc0 = {0.f, 0.f, 0.f, 0.f, 0.f, 0.f, 0.f, 0.f};
        v8f acc1 = {0.f, 0.f, 0.f, 0.f, 0.f, 0.f, 0.f, 0.f};

        #pragma unroll
        for (int kc = 0; kc < 4; ++kc) {
            const size_t ko = (size_t)kc * 32 * 16;
            v16bf bhi0 = *(const v16bf*)(hb0 + ko);
            v16bf blo0 = *(const v16bf*)(lb0 + ko);
            v16bf bhi1 = *(const v16bf*)(hb1 + ko);
            v16bf blo1 = *(const v16bf*)(lb1 + ko);
            // interleave the two independent accumulator chains
            acc0 = __builtin_amdgcn_wmma_f32_16x16x32_bf16(false, ahi[kc], false, bhi0,
                                                           (short)0, acc0, false, false);
            acc1 = __builtin_amdgcn_wmma_f32_16x16x32_bf16(false, ahi[kc], false, bhi1,
                                                           (short)0, acc1, false, false);
            acc0 = __builtin_amdgcn_wmma_f32_16x16x32_bf16(false, ahi[kc], false, blo0,
                                                           (short)0, acc0, false, false);
            acc1 = __builtin_amdgcn_wmma_f32_16x16x32_bf16(false, ahi[kc], false, blo1,
                                                           (short)0, acc1, false, false);
            acc0 = __builtin_amdgcn_wmma_f32_16x16x32_bf16(false, alo[kc], false, bhi0,
                                                           (short)0, acc0, false, false);
            acc1 = __builtin_amdgcn_wmma_f32_16x16x32_bf16(false, alo[kc], false, bhi1,
                                                           (short)0, acc1, false, false);
        }

        // h(d) = log((d+1)/(d+eps)) strictly decreasing -> max_n h == h(min_n d2)
        float mindA = INFINITY, mindB = INFINITY;
        #pragma unroll
        for (int r = 0; r < 8; ++r) {
            float dA = fmaxf(fmaf(-2.0f, acc0[r], zsr[r] + psqA), 0.0f);
            float dB = fmaxf(fmaf(-2.0f, acc1[r], zsr[r] + psqB), 0.0f);
            mindA = fminf(mindA, dA);
            mindB = fminf(mindB, dB);
        }
        mindA = fminf(mindA, __shfl_xor(mindA, 16, 32));
        mindB = fminf(mindB, __shfl_xor(mindB, 16, 32));
        if (lane < 16) {
            float dd = sqrtf(mindA);
            float h  = logf((dd + 1.0f) / (dd + EPS_));   // > 0 always
            atomicMax((int*)(out + (size_t)b * P_ + pA + lane), __float_as_int(h));
            dd = sqrtf(mindB);
            h  = logf((dd + 1.0f) / (dd + EPS_));
            atomicMax((int*)(out + (size_t)b * P_ + pB + lane), __float_as_int(h));
        }
    }
}

// ---------------------------------------------------------------------------
// Fallback (no/undersized workspace): converts B on the fly.
__global__ __launch_bounds__(256) void proto_sim_fallback(const float* __restrict__ z,
                                                          const float* __restrict__ proto,
                                                          float* __restrict__ out) {
    __shared__ float zt[D_ * 16];
    __shared__ float zsq[16];

    const int t      = threadIdx.x;
    const int lane   = t & 31;
    const int wid    = t >> 5;
    const int lanehi = lane >> 4;
    const int lanelo = lane & 15;

    const int ntile = blockIdx.x;
    const int b     = blockIdx.y;
    const int n0    = ntile * 16;

    const float* zb = z + (size_t)b * D_ * N_;
    #pragma unroll
    for (int j = 0; j < 8; ++j) {
        int i = t + 256 * j;
        int d = i >> 4, m = i & 15;
        int n = n0 + m;
        zt[i] = (n < N_) ? zb[d * N_ + n] : 0.0f;
    }
    __syncthreads();

    if (t < 16) {
        float s = 0.0f;
        for (int d = 0; d < D_; ++d) { float v = zt[d * 16 + t]; s += v * v; }
        zsq[t] = s;
    }

    v16bf ahi[4], alo[4];
    #pragma unroll
    for (int kc = 0; kc < 4; ++kc) {
        #pragma unroll
        for (int e = 0; e < 16; ++e) {
            int K  = e + 16 * lanehi;
            float f = zt[(32 * kc + K) * 16 + lanelo];
            bf16_t h = (bf16_t)f;
            ahi[kc][e] = h;
            alo[kc][e] = (bf16_t)(f - (float)h);
        }
    }
    __syncthreads();

    float zsr[8];
    #pragma unroll
    for (int r = 0; r < 8; ++r) {
        int nn = n0 + r + 8 * lanehi;
        zsr[r] = (nn < N_) ? zsq[r + 8 * lanehi] : INFINITY;
    }

    for (int pt = wid; pt < PT_; pt += 8) {
        const int p0 = pt * 16;
        const float* bp = proto + (size_t)(p0 + lanelo) * D_ + 16 * lanehi;

        v8f acc = {0.f, 0.f, 0.f, 0.f, 0.f, 0.f, 0.f, 0.f};
        float psq_part = 0.0f;

        #pragma unroll
        for (int kc = 0; kc < 4; ++kc) {
            v16bf bhi, blo;
            const float4* bp4 = (const float4*)(bp + 32 * kc);
            #pragma unroll
            for (int q = 0; q < 4; ++q) {
                float4 f4 = bp4[q];
                float fs[4] = {f4.x, f4.y, f4.z, f4.w};
                #pragma unroll
                for (int c = 0; c < 4; ++c) {
                    float f = fs[c];
                    bf16_t h = (bf16_t)f;
                    bhi[q * 4 + c] = h;
                    blo[q * 4 + c] = (bf16_t)(f - (float)h);
                    psq_part += f * f;
                }
            }
            acc = __builtin_amdgcn_wmma_f32_16x16x32_bf16(false, ahi[kc], false, bhi,
                                                          (short)0, acc, false, false);
            acc = __builtin_amdgcn_wmma_f32_16x16x32_bf16(false, ahi[kc], false, blo,
                                                          (short)0, acc, false, false);
            acc = __builtin_amdgcn_wmma_f32_16x16x32_bf16(false, alo[kc], false, bhi,
                                                          (short)0, acc, false, false);
        }

        float psq = psq_part + __shfl_xor(psq_part, 16, 32);
        float mind2 = INFINITY;
        #pragma unroll
        for (int r = 0; r < 8; ++r) {
            float d2 = fmaxf(fmaf(-2.0f, acc[r], zsr[r] + psq), 0.0f);
            mind2 = fminf(mind2, d2);
        }
        mind2 = fminf(mind2, __shfl_xor(mind2, 16, 32));
        if (lane < 16) {
            float dd = sqrtf(mind2);
            float h  = logf((dd + 1.0f) / (dd + EPS_));
            atomicMax((int*)(out + (size_t)b * P_ + p0 + lane), __float_as_int(h));
        }
    }
}

__global__ void zero_out_kernel(float* out, int n) {
    int i = blockIdx.x * blockDim.x + threadIdx.x;
    if (i < n) out[i] = 0.0f;
}

// ---------------------------------------------------------------------------
extern "C" void kernel_launch(void* const* d_in, const int* in_sizes, int n_in,
                              void* d_out, int out_size, void* d_ws, size_t ws_size,
                              hipStream_t stream) {
    const float* z = (const float*)d_in[0];
    const float* p = (const float*)d_in[1];
    float* out = (float*)d_out;

    zero_out_kernel<<<(out_size + 255) / 256, 256, 0, stream>>>(out, out_size);

    dim3 grid(NT_, B_);
    if (d_ws != nullptr && ws_size >= WS_NEEDED) {
        float*  psq = (float*)d_ws;
        bf16_t* whi = (bf16_t*)((char*)d_ws + PSQ_BYTES);
        bf16_t* wlo = whi + FRAG_ELEMS;

        proto_psq_kernel <<<(P_ + 255) / 256, 256, 0, stream>>>(p, psq);
        proto_frag_kernel<<<(PT_ * 4 + 7) / 8, 256, 0, stream>>>(p, whi, wlo);
        proto_sim_fast   <<<grid, 256, 0, stream>>>(z, psq, whi, wlo, out);
    } else {
        proto_sim_fallback<<<grid, 256, 0, stream>>>(z, p, out);
    }
}
